// GRUEncoder_4372276707792
// MI455X (gfx1250) — matrix operand order repaired and checked by reading, not compiled, not run on hardware
//
#include <hip/hip_runtime.h>
#include <math.h>

// ---------------- CDNA5 WMMA types ----------------
typedef __attribute__((ext_vector_type(16))) __bf16 v16bf;
typedef __attribute__((ext_vector_type(8)))  float  v8f;
typedef __attribute__((ext_vector_type(4)))  float  v4f;

#define B_  32
#define T_  512
#define H_  1024
#define H3  3072
#define KT  (H_/32)      // 32 k-tiles of 32
#define MROWS (B_*T_)    // 16384
#define MT  (MROWS/16)   // 1024 m-tiles
#define NT  (H3/16)      // 192 n-tiles

#define WMMA_BF16(a, b, c) \
    __builtin_amdgcn_wmma_f32_16x16x32_bf16(false, (a), false, (b), (short)0, (c), false, false)

__device__ __forceinline__ unsigned short f2bf(float f) {
    __bf16 h = (__bf16)f;
    return __builtin_bit_cast(unsigned short, h);
}

// ---------------------------------------------------------------------------
// Pack A: gather embedding rows for each (b,t) token, convert f32->bf16, and
// store in WMMA A-fragment layout (ISA 7.12.2, 16-bit A 16x32):
//   lane = r + half*16 holds, for element i (0..15): K = (i/8)*16 + half*8 + i%8
// Block (mTile,kTile) = 1024 bytes, layout [mTile][kTile] contiguous in k.
// ---------------------------------------------------------------------------
__global__ __launch_bounds__(256) void pack_a(const int* __restrict__ x,
                                              const float* __restrict__ emb,
                                              unsigned short* __restrict__ Ap) {
    int tid = blockIdx.x * blockDim.x + threadIdx.x;   // MROWS * 128 threads
    int m   = tid >> 7;
    int sub = tid & 127;           // kTile(32) * half(2) * g(2)
    int kTile = sub >> 2;
    int hf    = (sub >> 1) & 1;
    int g     = sub & 1;
    int token = x[m];
    const float* src = emb + (size_t)token * H_ + kTile * 32 + g * 16 + hf * 8;
    int r = m & 15, mTile = m >> 4;
    unsigned short* dst = Ap + ((size_t)mTile * KT + kTile) * 512
                             + (r + hf * 16) * 16 + g * 8;
    #pragma unroll
    for (int j = 0; j < 8; ++j) dst[j] = f2bf(src[j]);
}

// ---------------------------------------------------------------------------
// Pack W [H][3H] row-major -> bf16 B-fragment layout (32x16 tile):
//   lane = l + half*16 holds element i: K = (i/8)*16 + half*8 + i%8, N = l
// Block layout [nTile][kTile] (contiguous in k for the GEMM inner loop).
// ---------------------------------------------------------------------------
__global__ __launch_bounds__(256) void pack_w(const float* __restrict__ W,
                                              unsigned short* __restrict__ Wp) {
    int tid   = blockIdx.x * blockDim.x + threadIdx.x;  // NT*KT*32 threads
    int lane  = tid & 31;
    int kTile = (tid >> 5) & (KT - 1);
    int nTile = tid >> 10;
    int hf = lane >> 4, l = lane & 15;
    int n     = nTile * 16 + l;
    int kbase = kTile * 32 + hf * 8;
    unsigned short* dst = Wp + ((size_t)nTile * KT + kTile) * 512 + lane * 16;
    #pragma unroll
    for (int i = 0; i < 16; ++i) dst[i] = f2bf(W[(size_t)(kbase + i + (i >> 3) * 8) * H3 + n]);
}

// ---------------------------------------------------------------------------
// Big GEMM: gx[16384][3072] = Apack @ Wpack + b_i   (bf16 x bf16 -> f32)
// One wave computes a 32x64 strip: 2 M-tiles x 4 N-tiles (8 WMMAs per K-step
// sharing 6 fragment loads), register double-buffered so next-K loads overlap
// the current WMMA chain.
// ---------------------------------------------------------------------------
__global__ __launch_bounds__(256) void gemm_gx(const unsigned short* __restrict__ Ap,
                                               const unsigned short* __restrict__ Wp,
                                               const float* __restrict__ biasI,
                                               float* __restrict__ gx) {
    int wid  = blockIdx.x * 8 + (threadIdx.x >> 5);   // 24576 waves
    int lane = threadIdx.x & 31;
    int mPair  = wid / 48;           // 0..511 -> mTiles 2*mPair, 2*mPair+1
    int nGroup = wid % 48;           // 4 n-tiles each
    const v16bf* A0 = (const v16bf*)(Ap + (size_t)(mPair * 2) * KT * 512);
    const v16bf* A1 = A0 + (size_t)KT * 32;
    const v16bf* B0 = (const v16bf*)(Wp + (size_t)(nGroup * 4 + 0) * KT * 512);
    const v16bf* B1 = (const v16bf*)(Wp + (size_t)(nGroup * 4 + 1) * KT * 512);
    const v16bf* B2 = (const v16bf*)(Wp + (size_t)(nGroup * 4 + 2) * KT * 512);
    const v16bf* B3 = (const v16bf*)(Wp + (size_t)(nGroup * 4 + 3) * KT * 512);

    v8f c[2][4] = {};
    // prologue: fragments for k = 0
    v16bf a0 = A0[lane], a1 = A1[lane];
    v16bf b0 = B0[lane], b1 = B1[lane], b2 = B2[lane], b3 = B3[lane];

    for (int k = 0; k < KT; ++k) {
        int off = ((k + 1 < KT) ? (k + 1) : k) * 32 + lane;
        // issue next-iteration loads before consuming current fragments
        v16bf a0n = A0[off], a1n = A1[off];
        v16bf b0n = B0[off], b1n = B1[off], b2n = B2[off], b3n = B3[off];

        c[0][0] = WMMA_BF16(a0, b0, c[0][0]);
        c[0][1] = WMMA_BF16(a0, b1, c[0][1]);
        c[0][2] = WMMA_BF16(a0, b2, c[0][2]);
        c[0][3] = WMMA_BF16(a0, b3, c[0][3]);
        c[1][0] = WMMA_BF16(a1, b0, c[1][0]);
        c[1][1] = WMMA_BF16(a1, b1, c[1][1]);
        c[1][2] = WMMA_BF16(a1, b2, c[1][2]);
        c[1][3] = WMMA_BF16(a1, b3, c[1][3]);

        a0 = a0n; a1 = a1n; b0 = b0n; b1 = b1n; b2 = b2n; b3 = b3n;
    }

    int hf = lane >> 4, l = lane & 15;
    #pragma unroll
    for (int mi = 0; mi < 2; ++mi) {
        int rowBase = (mPair * 2 + mi) * 16 + 8 * hf;   // C layout: M = r + 8*half
        #pragma unroll
        for (int j = 0; j < 4; ++j) {
            int n = (nGroup * 4 + j) * 16 + l;          // N = lane%16
            float bi = biasI[n];
            float* colp = gx + (size_t)rowBase * H3 + n;
            #pragma unroll
            for (int r = 0; r < 8; ++r) colp[(size_t)r * H3] = c[mi][j][r] + bi;
        }
    }
}

// ---------------------------------------------------------------------------
// One recurrent step. 128 waves: mTile in {0,1} (batch halves), hTile in 0..63
// (16 hidden columns). Each wave computes the z/r/n tiles of gh = h @ U,
// register double-buffered, then fuses the GRU gate update + padding mask.
// ---------------------------------------------------------------------------
__global__ __launch_bounds__(64) void scan_step(const float* __restrict__ hprev,
                                                int hstride,
                                                const unsigned short* __restrict__ Wrp,
                                                const float* __restrict__ gx,
                                                const float* __restrict__ biasR,
                                                const int* __restrict__ x,
                                                float* __restrict__ out,
                                                int t) {
    int wid  = blockIdx.x * 2 + (threadIdx.x >> 5);   // 128 waves total
    int lane = threadIdx.x & 31;
    int mTile = wid >> 6;         // 0..1
    int hTile = wid & 63;         // 0..63
    int hf = lane >> 4, l = lane & 15;

    const float* hrow = hprev + (size_t)(mTile * 16 + l) * hstride;  // A row = lane%16
    const v16bf* Bz = (const v16bf*)(Wrp + (size_t)(0 * 64 + hTile) * KT * 512);
    const v16bf* Br = (const v16bf*)(Wrp + (size_t)(1 * 64 + hTile) * KT * 512);
    const v16bf* Bn = (const v16bf*)(Wrp + (size_t)(2 * 64 + hTile) * KT * 512);

    v8f az = {}, ar = {}, an = {};
    // prologue loads for k = 0
    v4f f0 = *(const v4f*)(hrow + hf * 8);
    v4f f1 = *(const v4f*)(hrow + hf * 8 + 4);
    v4f f2 = *(const v4f*)(hrow + 16 + hf * 8);
    v4f f3 = *(const v4f*)(hrow + 16 + hf * 8 + 4);
    v16bf bz = Bz[lane], br = Br[lane], bn = Bn[lane];

    for (int k = 0; k < KT; ++k) {
        int kn  = (k + 1 < KT) ? (k + 1) : k;
        int k0  = kn * 32 + hf * 8;
        int off = kn * 32 + lane;
        // next-iteration loads in flight while we convert + WMMA the current
        v4f nf0 = *(const v4f*)(hrow + k0);
        v4f nf1 = *(const v4f*)(hrow + k0 + 4);
        v4f nf2 = *(const v4f*)(hrow + k0 + 16);
        v4f nf3 = *(const v4f*)(hrow + k0 + 16 + 4);
        v16bf nbz = Bz[off], nbr = Br[off], nbn = Bn[off];

        v16bf a;
        #pragma unroll
        for (int j = 0; j < 4; ++j) {
            a[j]      = (__bf16)f0[j];
            a[4 + j]  = (__bf16)f1[j];
            a[8 + j]  = (__bf16)f2[j];
            a[12 + j] = (__bf16)f3[j];
        }
        az = WMMA_BF16(a, bz, az);
        ar = WMMA_BF16(a, br, ar);
        an = WMMA_BF16(a, bn, an);

        f0 = nf0; f1 = nf1; f2 = nf2; f3 = nf3;
        bz = nbz; br = nbr; bn = nbn;
    }

    int col = hTile * 16 + l;
    float brz = biasR[col], brr = biasR[1024 + col], brn = biasR[2048 + col];
    int bBase = mTile * 16 + 8 * hf;     // C layout: M = r + 8*half
    #pragma unroll
    for (int r = 0; r < 8; ++r) {
        int b = bBase + r;
        int token = x[b * T_ + t];
        const float* gxr = gx + ((size_t)b * T_ + t) * H3;
        float hp = hprev[(size_t)b * hstride + col];
        float xz = gxr[col], xr = gxr[1024 + col], xn = gxr[2048 + col];
        float hz = az[r] + brz, hr = ar[r] + brr, hn = an[r] + brn;
        float z  = 1.0f / (1.0f + __expf(-(xz + hz)));
        float rr = 1.0f / (1.0f + __expf(-(xr + hr)));
        float hh = tanhf(xn + rr * hn);
        float hnew = z * hp + (1.0f - z) * hh;
        if (token == 0) hnew = hp;       // mask_zero: carry state
        out[((size_t)b * T_ + t) * H_ + col] = hnew;
    }
}

__global__ __launch_bounds__(256) void final_state(const float* __restrict__ out,
                                                   float* __restrict__ state) {
    int i = blockIdx.x * blockDim.x + threadIdx.x;   // B_*H_ threads
    int b = i >> 10, col = i & (H_ - 1);
    state[i] = out[((size_t)b * T_ + (T_ - 1)) * H_ + col];
}

// ---------------------------------------------------------------------------
extern "C" void kernel_launch(void* const* d_in, const int* in_sizes, int n_in,
                              void* d_out, int out_size, void* d_ws, size_t ws_size,
                              hipStream_t stream) {
    const int*   x      = (const int*)d_in[0];
    const float* hidden = (const float*)d_in[1];
    const float* emb    = (const float*)d_in[2];
    const float* Wk     = (const float*)d_in[3];
    const float* Wr     = (const float*)d_in[4];
    const float* bias   = (const float*)d_in[5];    // [2,3H]
    float* out   = (float*)d_out;                   // [B,T,H]
    float* state = out + (size_t)B_ * T_ * H_;      // [B,H]

    char* ws = (char*)d_ws;
    float* gx = (float*)ws;                                           // 192 MB
    unsigned short* Ap  = (unsigned short*)(ws + (size_t)MROWS * H3 * 4);   // 32 MB
    unsigned short* Wkp = (unsigned short*)((char*)Ap + (size_t)MROWS * H_ * 2);
    unsigned short* Wrp = (unsigned short*)((char*)Wkp + (size_t)H_ * H3 * 2);

    pack_a<<<(MROWS * 128) / 256, 256, 0, stream>>>(x, emb, Ap);
    pack_w<<<(NT * KT * 32) / 256, 256, 0, stream>>>(Wk, Wkp);
    pack_w<<<(NT * KT * 32) / 256, 256, 0, stream>>>(Wr, Wrp);
    gemm_gx<<<(MT / 2) * 48 / 8, 256, 0, stream>>>(Ap, Wkp, bias, gx);

    for (int t = 0; t < T_; ++t) {
        const float* hprev = (t == 0) ? hidden : (out + (size_t)(t - 1) * H_);
        int hstride = (t == 0) ? H_ : (T_ * H_);
        scan_step<<<64, 64, 0, stream>>>(hprev, hstride, Wrp, gx, bias + H3, x, out, t);
    }
    final_state<<<(B_ * H_) / 256, 256, 0, stream>>>(out, state);
}
